// LuGTP_13804024889770
// MI455X (gfx1250) — matrix operand order repaired and checked by reading, not compiled, not run on hardware
//
#include <hip/hip_runtime.h>

// ---------------------------------------------------------------------------
// Types for CDNA5 WMMA
// ---------------------------------------------------------------------------
typedef __attribute__((ext_vector_type(16))) __bf16        v16bf;
typedef __attribute__((ext_vector_type(2)))  __bf16        v2bf;
typedef __attribute__((ext_vector_type(8)))  float         v8f;
typedef __attribute__((ext_vector_type(8)))  unsigned int  v8u;

static constexpr int GB = 64;      // batch
static constexpr int GE = 16384;   // edges per graph

// Native f32 -> bf16 (RNE); backend picks hw cvt where available.
__device__ __forceinline__ unsigned short f32_to_bf16(float f) {
  __bf16 b = (__bf16)f;
  return __builtin_bit_cast(unsigned short, b);
}

__device__ __forceinline__ unsigned int pack_bf16x2(float lo, float hi) {
  v2bf p;
  p[0] = (__bf16)lo;
  p[1] = (__bf16)hi;
  return __builtin_bit_cast(unsigned int, p);
}

// ---------------------------------------------------------------------------
// Utility fills
// ---------------------------------------------------------------------------
__global__ void zero_i32_kernel(int* __restrict__ p, long n) {
  long t = (long)blockIdx.x * blockDim.x + threadIdx.x;
  if (t < n) p[t] = 0;
}

// ---------------------------------------------------------------------------
// CSR build: counts -> per-batch exclusive scan -> bucket fill
// ---------------------------------------------------------------------------
__global__ void count_kernel(const int* __restrict__ src, const int* __restrict__ dst,
                             const unsigned char* __restrict__ mask,
                             int* __restrict__ cnt_out, int* __restrict__ cnt_in,
                             int N, long total) {
  long t = (long)blockIdx.x * blockDim.x + threadIdx.x;
  if (t >= total) return;
  if (mask && !mask[t]) return;
  int b = (int)(t / GE);
  atomicAdd(&cnt_out[(long)b * N + src[t]], 1);
  atomicAdd(&cnt_in [(long)b * N + dst[t]], 1);
}

__global__ void norm_from_cnt_kernel(const int* __restrict__ cnt, float* __restrict__ nrm, long n) {
  long t = (long)blockIdx.x * blockDim.x + threadIdx.x;
  if (t < n) nrm[t] = rsqrtf(fmaxf((float)cnt[t], 1.0f));
}

// One block per batch, N threads (N power of two <= 1024): exclusive scan of
// cnt_in into offs and cursor (Hillis-Steele in LDS).
__global__ __launch_bounds__(1024)
void scan_kernel(const int* __restrict__ cnt, int* __restrict__ offs,
                 int* __restrict__ cursor, int N) {
  __shared__ int buf[1024];
  const int b = blockIdx.x;
  const int t = threadIdx.x;
  int v = cnt[(long)b * N + t];
  buf[t] = v;
  __syncthreads();
  for (int off = 1; off < N; off <<= 1) {
    int x = (t >= off) ? buf[t - off] : 0;
    __syncthreads();
    buf[t] += x;
    __syncthreads();
  }
  int excl = buf[t] - v;
  offs  [(long)b * N + t] = excl;
  cursor[(long)b * N + t] = excl;
}

__global__ void fill_elist_kernel(const int* __restrict__ src, const int* __restrict__ dst,
                                  const unsigned char* __restrict__ mask,
                                  int* __restrict__ cursor, int* __restrict__ elist,
                                  int N, long total) {
  long t = (long)blockIdx.x * blockDim.x + threadIdx.x;
  if (t >= total) return;
  if (mask && !mask[t]) return;
  int b = (int)(t / GE);
  int slot = atomicAdd(&cursor[(long)b * N + dst[t]], 1);
  elist[(long)b * GE + slot] = src[t];
}

// ---------------------------------------------------------------------------
// bf16 WMMA GEMM:  C[Mtot,Nc] = rowscale[row] * (A[map(row),:K] @ W[K,Nc])
// block = 256 threads (8 waves); block tile 128(M) x 64(N); K step 32.
// grid: x = N tiles (fast) so consecutive blocks share the same A slab in L2.
// ---------------------------------------------------------------------------
template <bool HAS_MAP>
__global__ __launch_bounds__(256)
void wmma_gemm_kernel(const float* __restrict__ A, const int* __restrict__ rowmap,
                      const float* __restrict__ W, const float* __restrict__ rowscale,
                      float* __restrict__ C, int K, int Nc) {
  __shared__ unsigned short Asm[128][36];  // [row][k]; stride 36 keeps b64 alignment
  __shared__ unsigned short Bsm[64][36];   // [col][k]

  const int tid  = threadIdx.x;
  const int wave = tid >> 5;               // 0..7  -> M strip
  const int lane = tid & 31;
  const int ln   = lane & 15;
  const int hi   = lane >> 4;

  const int m0 = blockIdx.y * 128;
  const int n0 = blockIdx.x * 64;

  const int ar  = tid >> 3;
  const int akq = (tid & 7) * 4;
  const int bk  = tid >> 4;
  const int bj4 = (tid & 15) * 4;

  const float* aptr[4];
  #pragma unroll
  for (int it = 0; it < 4; ++it) {
    int grow = m0 + ar + 32 * it;
    long arow = HAS_MAP ? (long)rowmap[grow] : (long)grow;
    aptr[it] = A + arow * (long)K + akq;
  }
  const float* wptr[2];
  #pragma unroll
  for (int it = 0; it < 2; ++it)
    wptr[it] = W + (long)(bk + 16 * it) * Nc + n0 + bj4;

  v8f zero;
  #pragma unroll
  for (int i = 0; i < 8; ++i) zero[i] = 0.0f;
  v8f acc[4];
  #pragma unroll
  for (int i = 0; i < 4; ++i) acc[i] = zero;

  float4 aval[4], bval[2];
  #pragma unroll
  for (int it = 0; it < 4; ++it) aval[it] = *(const float4*)(aptr[it]);
  #pragma unroll
  for (int it = 0; it < 2; ++it) bval[it] = *(const float4*)(wptr[it]);

  for (int k0 = 0; k0 < K; k0 += 32) {
    #pragma unroll
    for (int it = 0; it < 4; ++it) {
      uint2 pk;
      pk.x = pack_bf16x2(aval[it].x, aval[it].y);
      pk.y = pack_bf16x2(aval[it].z, aval[it].w);
      *(uint2*)&Asm[ar + 32 * it][akq] = pk;
    }
    #pragma unroll
    for (int it = 0; it < 2; ++it) {
      int kk = bk + 16 * it;
      Bsm[bj4 + 0][kk] = f32_to_bf16(bval[it].x);
      Bsm[bj4 + 1][kk] = f32_to_bf16(bval[it].y);
      Bsm[bj4 + 2][kk] = f32_to_bf16(bval[it].z);
      Bsm[bj4 + 3][kk] = f32_to_bf16(bval[it].w);
    }
    __syncthreads();

    if (k0 + 32 < K) {
      #pragma unroll
      for (int it = 0; it < 4; ++it)
        aval[it] = *(const float4*)(aptr[it] + (k0 + 32));
      #pragma unroll
      for (int it = 0; it < 2; ++it)
        bval[it] = *(const float4*)(wptr[it] + (long)(k0 + 32) * Nc);
    }

    // A fragment (ISA layout): vgpr v<4: K=2v,2v+1 (+8 hi); v>=4: K=16+2(v-4) (+8 hi)
    v8u au;
    const unsigned short* ap = &Asm[wave * 16 + ln][0];
    #pragma unroll
    for (int v = 0; v < 8; ++v) {
      int kb = ((v < 4) ? (2 * v) : (16 + 2 * (v - 4))) + 8 * hi;
      au[v] = *(const unsigned int*)(ap + kb);
    }
    v16bf afrag = __builtin_bit_cast(v16bf, au);

    #pragma unroll
    for (int nt = 0; nt < 4; ++nt) {
      // B fragment: col = lane%16; v<4: K=2v,2v+1 (+8 hi); v>=4: K=16+2(v-4) (+8 hi)
      v8u bu;
      const unsigned short* bp = &Bsm[nt * 16 + ln][0];
      #pragma unroll
      for (int v = 0; v < 8; ++v) {
        int kb = ((v < 4) ? (2 * v) : (16 + 2 * (v - 4))) + 8 * hi;
        bu[v] = *(const unsigned int*)(bp + kb);
      }
      v16bf bfrag = __builtin_bit_cast(v16bf, bu);
      acc[nt] = __builtin_amdgcn_wmma_f32_16x16x32_bf16(
          false, afrag, false, bfrag, (short)0, acc[nt], false, false);
    }
    __syncthreads();
  }

  #pragma unroll
  for (int nt = 0; nt < 4; ++nt) {
    int col = n0 + nt * 16 + ln;
    #pragma unroll
    for (int g = 0; g < 8; ++g) {
      int row = m0 + wave * 16 + g + 8 * hi;
      float v = acc[nt][g];
      if (rowscale) v *= rowscale[row];
      C[(long)row * Nc + col] = v;
    }
  }
}

// ---------------------------------------------------------------------------
// Gather aggregation (no float atomics): one block per (b, dst-node).
// outc[bn,:] = relu( (sum_{e in in(bn)} h[b, src_e, :]) * n_dst[bn] + bias )
// ---------------------------------------------------------------------------
__global__ __launch_bounds__(128)
void gather_conv_kernel(const float* __restrict__ h, const int* __restrict__ elist,
                        const int* __restrict__ offs, const int* __restrict__ cnt,
                        const float* __restrict__ ndst, const float* __restrict__ bias,
                        float* __restrict__ outc, int N, int D) {
  const int bn = blockIdx.x;
  const int b  = bn / N;
  const int t  = threadIdx.x;
  const int c  = t * 4;
  if (c >= D) return;
  const int  deg  = cnt[bn];
  const long base = (long)b * GE + offs[bn];
  const float* hb = h + (long)b * N * D + c;
  float4 s = make_float4(0.f, 0.f, 0.f, 0.f);
  for (int e = 0; e < deg; ++e) {
    int sn = elist[base + e];
    const float4 v = *(const float4*)(hb + (long)sn * D);
    s.x += v.x; s.y += v.y; s.z += v.z; s.w += v.w;
  }
  float nd = ndst[bn];
  float4 o;
  o.x = fmaxf(s.x * nd + bias[c + 0], 0.f);
  o.y = fmaxf(s.y * nd + bias[c + 1], 0.f);
  o.z = fmaxf(s.z * nd + bias[c + 2], 0.f);
  o.w = fmaxf(s.w * nd + bias[c + 3], 0.f);
  *(float4*)(outc + (long)bn * D + c) = o;
}

// ---------------------------------------------------------------------------
// Pool scoring: 1/||p|| then score[r] = (x[r,:] . p) * inv_norm (wave per row)
// ---------------------------------------------------------------------------
__global__ void pnorm_kernel(const float* __restrict__ p, int D, float* __restrict__ out) {
  __shared__ float red[256];
  float s = 0.0f;
  for (int i = threadIdx.x; i < D; i += 256) s += p[i] * p[i];
  red[threadIdx.x] = s;
  __syncthreads();
  for (int off = 128; off > 0; off >>= 1) {
    if ((int)threadIdx.x < off) red[threadIdx.x] += red[threadIdx.x + off];
    __syncthreads();
  }
  if (threadIdx.x == 0) out[0] = rsqrtf(red[0]);
}

__global__ void score_kernel(const float* __restrict__ x, const float* __restrict__ p,
                             const float* __restrict__ inv_norm, float* __restrict__ score,
                             int D, long rows) {
  long r = ((long)blockIdx.x * blockDim.x + threadIdx.x) >> 5;
  int lane = threadIdx.x & 31;
  if (r >= rows) return;
  const float* xp = x + r * D;
  float s = 0.0f;
  for (int i = lane; i < D; i += 32) s += xp[i] * p[i];
  #pragma unroll
  for (int off = 16; off > 0; off >>= 1) s += __shfl_xor(s, off, 32);
  if (lane == 0) score[r] = s * inv_norm[0];
}

// ---------------------------------------------------------------------------
// Per-batch top-k via in-LDS bitonic sort (N power of two, <= 1024).
// ---------------------------------------------------------------------------
__global__ __launch_bounds__(1024)
void topk_kernel(const float* __restrict__ score, int N, int k,
                 int* __restrict__ pos, int* __restrict__ rowmap, float* __restrict__ gate) {
  __shared__ float key[1024];
  __shared__ int   idx[1024];
  const int b = blockIdx.x;
  const int t = threadIdx.x;
  key[t] = score[(long)b * N + t];
  idx[t] = t;
  pos[(long)b * N + t] = -1;
  __syncthreads();
  for (int sz = 2; sz <= N; sz <<= 1) {
    for (int j = sz >> 1; j > 0; j >>= 1) {
      int ixj = t ^ j;
      if (ixj > t) {
        bool descend = ((t & sz) == 0);
        float k0 = key[t], k1 = key[ixj];
        bool sw = descend ? (k0 < k1) : (k0 > k1);
        if (sw) {
          key[t] = k1; key[ixj] = k0;
          int tmp = idx[t]; idx[t] = idx[ixj]; idx[ixj] = tmp;
        }
      }
      __syncthreads();
    }
  }
  if (t < k) {
    float v = key[t];
    int   n = idx[t];
    rowmap[(long)b * k + t] = b * N + n;
    gate  [(long)b * k + t] = 1.0f / (1.0f + __expf(-v));
    pos[(long)b * N + n] = t;
  }
}

// Remap edges into pooled node space; invalid edges -> (0,0,mask=0)
__global__ void remap_kernel(const int* __restrict__ src, const int* __restrict__ dst,
                             const unsigned char* __restrict__ mask_in, const int* __restrict__ pos,
                             int N, int* __restrict__ s_out, int* __restrict__ d_out,
                             unsigned char* __restrict__ m_out, long total) {
  long t = (long)blockIdx.x * blockDim.x + threadIdx.x;
  if (t >= total) return;
  int b  = (int)(t / GE);
  int ns = pos[(long)b * N + src[t]];
  int nd = pos[(long)b * N + dst[t]];
  bool valid = (mask_in ? (mask_in[t] != 0) : true) && ns >= 0 && nd >= 0;
  s_out[t] = valid ? ns : 0;
  d_out[t] = valid ? nd : 0;
  m_out[t] = valid ? 1 : 0;
}

// Readout: mean + max over nodes into out[b, off .. off+2D)
__global__ void readout_kernel(const float* __restrict__ x, int n, int D,
                               float* __restrict__ out, int out_stride, int off, long total) {
  long t = (long)blockIdx.x * blockDim.x + threadIdx.x;
  if (t >= total) return;
  int b = (int)(t / D);
  int c = (int)(t % D);
  const float* xp = x + (long)b * n * D + c;
  float s = 0.0f, mx = -3.402823466e38f;
  for (int i = 0; i < n; ++i) {
    float v = xp[(long)i * D];
    s += v;
    mx = fmaxf(mx, v);
  }
  out[(long)b * out_stride + off + c]     = s / (float)n;
  out[(long)b * out_stride + off + D + c] = mx;
}

// ---------------------------------------------------------------------------
// Host orchestration
// ---------------------------------------------------------------------------
static inline unsigned nblk(long n, int t) { return (unsigned)((n + t - 1) / t); }

static void launch_gemm(hipStream_t stream, const float* A, const int* rowmap,
                        const float* W, const float* rowscale, float* C,
                        long Mtot, int K, int Nc) {
  dim3 grid((unsigned)(Nc / 64), (unsigned)(Mtot / 128));
  if (rowmap)
    wmma_gemm_kernel<true><<<grid, 256, 0, stream>>>(A, rowmap, W, rowscale, C, K, Nc);
  else
    wmma_gemm_kernel<false><<<grid, 256, 0, stream>>>(A, nullptr, W, rowscale, C, K, Nc);
}

static void run_layer(hipStream_t stream,
                      const float* x_in, int N, int Din, int Dout,
                      const int* esrc, const int* edst, const unsigned char* emask,
                      const float* W, const float* bias, const float* p, const float* Wp,
                      int k, int Pout, float* x_out, float* out, int out_off,
                      float* h_buf, float* conv_buf,
                      float* nsrc, float* ndst, float* score, int* pos,
                      int* rowmap, float* gate, float* pnrm,
                      int* cnt_out, int* cnt_in, int* offs, int* cursor, int* elist,
                      int* s_out, int* d_out, unsigned char* m_out) {
  const long rows  = (long)GB * N;
  const long edges = (long)GB * GE;

  // CSR by destination + degree norms (int atomics only)
  zero_i32_kernel<<<nblk(rows, 256), 256, 0, stream>>>(cnt_out, rows);
  zero_i32_kernel<<<nblk(rows, 256), 256, 0, stream>>>(cnt_in, rows);
  count_kernel<<<nblk(edges, 256), 256, 0, stream>>>(esrc, edst, emask, cnt_out, cnt_in, N, edges);
  norm_from_cnt_kernel<<<nblk(rows, 256), 256, 0, stream>>>(cnt_out, nsrc, rows);
  norm_from_cnt_kernel<<<nblk(rows, 256), 256, 0, stream>>>(cnt_in, ndst, rows);
  scan_kernel<<<GB, N, 0, stream>>>(cnt_in, offs, cursor, N);
  fill_elist_kernel<<<nblk(edges, 256), 256, 0, stream>>>(esrc, edst, emask, cursor, elist, N, edges);

  // h = (x @ W) * n_src
  launch_gemm(stream, x_in, nullptr, W, nsrc, h_buf, rows, Din, Dout);

  // conv = relu(gathered-sum * n_dst + bias)   (no float atomics)
  gather_conv_kernel<<<(unsigned)rows, 128, 0, stream>>>(h_buf, elist, offs, cnt_in,
                                                         ndst, bias, conv_buf, N, Dout);

  // scoring + top-k
  pnorm_kernel<<<1, 256, 0, stream>>>(p, Dout, pnrm);
  score_kernel<<<nblk(rows * 32, 256), 256, 0, stream>>>(conv_buf, p, pnrm, score, Dout, rows);
  topk_kernel<<<GB, N, 0, stream>>>(score, N, k, pos, rowmap, gate);

  // x_out = sigmoid(vals) * (x_sel @ Wp)   (gather + gate fused into GEMM)
  launch_gemm(stream, conv_buf, rowmap, Wp, gate, x_out, (long)GB * k, Dout, Pout);

  // readout straight into output slice
  {
    long total = (long)GB * Pout;
    readout_kernel<<<nblk(total, 256), 256, 0, stream>>>(x_out, k, Pout, out, 896, out_off, total);
  }

  // remap edges for next layer
  if (s_out) {
    remap_kernel<<<nblk(edges, 256), 256, 0, stream>>>(esrc, edst, emask, pos, N,
                                                       s_out, d_out, m_out, edges);
  }
}

extern "C" void kernel_launch(void* const* d_in, const int* in_sizes, int n_in,
                              void* d_out, int out_size, void* d_ws, size_t ws_size,
                              hipStream_t stream) {
  const float* feat = (const float*)d_in[0];
  const int*   src  = (const int*)d_in[1];
  const int*   dst  = (const int*)d_in[2];
  const float* W1 = (const float*)d_in[3];  const float* b1 = (const float*)d_in[4];
  const float* p1 = (const float*)d_in[5];  const float* Wp1 = (const float*)d_in[6];
  const float* W2 = (const float*)d_in[7];  const float* b2 = (const float*)d_in[8];
  const float* p2 = (const float*)d_in[9];  const float* Wp2 = (const float*)d_in[10];
  const float* W3 = (const float*)d_in[11]; const float* b3 = (const float*)d_in[12];
  const float* p3 = (const float*)d_in[13]; const float* Wp3 = (const float*)d_in[14];
  float* out = (float*)d_out;

  // Workspace carve (bump allocator, 256B aligned)
  char* w = (char*)d_ws;
  auto alloc = [&](size_t bytes) -> void* {
    void* r = (void*)w;
    w += (bytes + 255) & ~(size_t)255;
    return r;
  };
  const long BIGN = (long)GB * 1024 * 512;                   // 33.5M floats
  float* BIG0 = (float*)alloc(BIGN * 4);                     // h buffer
  float* BIG1 = (float*)alloc(BIGN * 4);                     // conv output
  float* x1   = (float*)alloc((long)GB * 512 * 256 * 4);
  float* x2   = (float*)alloc((long)GB * 256 * 128 * 4);
  float* x3   = (float*)alloc((long)GB * 128 * 64  * 4);
  int* e1s = (int*)alloc((long)GB * GE * 4);
  int* e1d = (int*)alloc((long)GB * GE * 4);
  unsigned char* e1m = (unsigned char*)alloc((long)GB * GE);
  int* e2s = (int*)alloc((long)GB * GE * 4);
  int* e2d = (int*)alloc((long)GB * GE * 4);
  unsigned char* e2m = (unsigned char*)alloc((long)GB * GE);
  float* nsrc   = (float*)alloc((long)GB * 1024 * 4);
  float* ndst   = (float*)alloc((long)GB * 1024 * 4);
  float* score  = (float*)alloc((long)GB * 1024 * 4);
  int*   pos    = (int*)  alloc((long)GB * 1024 * 4);
  int*   rowmap = (int*)  alloc((long)GB * 512 * 4);
  float* gate   = (float*)alloc((long)GB * 512 * 4);
  float* pnrm   = (float*)alloc(256);
  int* cnt_out = (int*)alloc((long)GB * 1024 * 4);
  int* cnt_in  = (int*)alloc((long)GB * 1024 * 4);
  int* offs    = (int*)alloc((long)GB * 1024 * 4);
  int* cursor  = (int*)alloc((long)GB * 1024 * 4);
  int* elist   = (int*)alloc((long)GB * GE * 4);

  // Layer 1: N=1024, 1024 -> conv 512 -> pool k=512, proj 256; readout -> out[:,0:512]
  run_layer(stream, feat, 1024, 1024, 512, src, dst, nullptr,
            W1, b1, p1, Wp1, 512, 256, x1, out, 0,
            BIG0, BIG1, nsrc, ndst, score, pos, rowmap, gate, pnrm,
            cnt_out, cnt_in, offs, cursor, elist,
            e1s, e1d, e1m);

  // Layer 2: N=512, 256 -> conv 256 -> pool k=256, proj 128; readout -> out[:,512:768]
  run_layer(stream, x1, 512, 256, 256, e1s, e1d, e1m,
            W2, b2, p2, Wp2, 256, 128, x2, out, 512,
            BIG0, BIG1, nsrc, ndst, score, pos, rowmap, gate, pnrm,
            cnt_out, cnt_in, offs, cursor, elist,
            e2s, e2d, e2m);

  // Layer 3: N=256, 128 -> conv 128 -> pool k=128, proj 64; readout -> out[:,768:896]
  run_layer(stream, x2, 256, 128, 128, e2s, e2d, e2m,
            W3, b3, p3, Wp3, 128, 64, x3, out, 768,
            BIG0, BIG1, nsrc, ndst, score, pos, rowmap, gate, pnrm,
            cnt_out, cnt_in, offs, cursor, elist,
            nullptr, nullptr, nullptr);
}
